// TPALSTM_28303834481025
// MI455X (gfx1250) — compile-verified, matrix operand order checked by reading, not run on hardware
//
#include <hip/hip_runtime.h>
#include <hip/hip_bf16.h>
#include <math.h>

// Problem dims (match reference)
constexpr int BB   = 256;   // batch
constexpr int TT   = 168;   // time steps
constexpr int FIN  = 32;    // input features
constexpr int HSZ  = 512;   // hidden size
constexpr int NLAY = 2;     // LSTM layers
constexpr int NOUT = 24;
constexpr int NFN  = 16;    // conv filters
constexpr int G4   = 4 * HSZ;   // 2048 gate columns
constexpr int KCAT = 2 * HSZ;   // 1024 fused K (input | hidden)

// LDS tiling for the LSTM step GEMM
constexpr int KT   = 64;             // k-tile
constexpr int LDT  = 80;             // padded row stride (halves): 160B, 16B-aligned
constexpr int A_EL = 32 * LDT;       // A tile elements per buffer
constexpr int B_EL = 128 * LDT;      // B tile elements per buffer
constexpr int SMEM_BYTES = 2 * (A_EL + B_EL) * 2;   // 51200 B double-buffered

typedef __attribute__((ext_vector_type(16))) _Float16 v16h;
typedef __attribute__((ext_vector_type(8)))  _Float16 v8h;
typedef __attribute__((ext_vector_type(8)))  float    v8f;

__device__ __forceinline__ float sigf(float x) { return 1.0f / (1.0f + __expf(-x)); }

// ---------------------------------------------------------------------------
// f32 -> f16 elementwise convert
__global__ void k_cvt_f16(const float* __restrict__ src, _Float16* __restrict__ dst, int n) {
    int i = blockIdx.x * blockDim.x + threadIdx.x;
    if (i < n) dst[i] = (_Float16)src[i];
}

// Build fused, gate-interleaved weight Wcat[l][n'][k], n' = grp*128 + gate*32 + u
__global__ void k_build_wcat(const float* __restrict__ Wih, const float* __restrict__ Whh,
                             _Float16* __restrict__ Wcat, int n) {
    int i = blockIdx.x * blockDim.x + threadIdx.x;
    if (i >= n) return;
    int l    = i / (G4 * KCAT);
    int rem  = i % (G4 * KCAT);
    int np   = rem / KCAT;
    int k    = rem % KCAT;
    int grp  = np >> 7;          // 0..15  (block of 32 hidden units)
    int gate = (np >> 5) & 3;    // i,f,g,o
    int u    = np & 31;
    int row  = gate * HSZ + grp * 32 + u;  // original gate-matrix row
    float v  = (k < HSZ) ? Wih[((size_t)l * G4 + row) * HSZ + k]
                         : Whh[((size_t)l * G4 + row) * HSZ + (k - HSZ)];
    Wcat[i] = (_Float16)v;
}

// Fused bias in the same permuted gate order: biascat[l][n'] = bih+bhh
__global__ void k_build_bias(const float* __restrict__ bih, const float* __restrict__ bhh,
                             float* __restrict__ biascat, int n) {
    int i = blockIdx.x * blockDim.x + threadIdx.x;
    if (i >= n) return;
    int l   = i / G4;
    int np  = i % G4;
    int grp = np >> 7, gate = (np >> 5) & 3, u = np & 31;
    int row = gate * HSZ + grp * 32 + u;
    biascat[i] = bih[l * G4 + row] + bhh[l * G4 + row];
}

// ---------------------------------------------------------------------------
// xconcat[t][b][:] = x[b][t][:] @ W_hid^T + b_hid   (M = T*B = 43008, N = 512, K = 32)
// block: 256 thr (8 waves), each wave: 16 rows x 64 cols; grid (336, 8)
__global__ void k_xc_gemm(const _Float16* __restrict__ x16,   // (B,T,FIN)
                          const _Float16* __restrict__ wh16,  // (HSZ,FIN)
                          const float* __restrict__ b_hid,
                          _Float16* __restrict__ xc16) {      // (T,B,HSZ)
    int wave = threadIdx.x >> 5;
    int lane = threadIdx.x & 31;
    int r  = lane & 15;
    int kh = lane >> 4;
    int m0 = blockIdx.x * 128 + wave * 16;
    int n0 = blockIdx.y * 64;

    int m  = m0 + r;                 // row = t*BB + b
    int tA = m >> 8, bA = m & 255;
    v16h a = *(const v16h*)(x16 + ((size_t)bA * TT + tA) * FIN + kh * 16);

    v8f acc[4];
#pragma unroll
    for (int j = 0; j < 4; ++j) {
        v16h bf = *(const v16h*)(wh16 + (size_t)(n0 + j * 16 + r) * FIN + kh * 16);
        v8f z = {};
        acc[j] = __builtin_amdgcn_wmma_f32_16x16x32_f16(false, a, false, bf,
                                                        (short)0, z, false, false);
    }
#pragma unroll
    for (int j = 0; j < 4; ++j) {
        int col = n0 + j * 16 + r;
        float bias = b_hid[col];
#pragma unroll
        for (int v = 0; v < 8; ++v) {
            int row = m0 + kh * 8 + v;
            xc16[(size_t)row * HSZ + col] = (_Float16)(acc[j][v] + bias);
        }
    }
}

// ---------------------------------------------------------------------------
// One LSTM step for one layer: LDS-staged double-buffered WMMA GEMM + fused cell update.
// gates(32 x 128 per block) = [xA|hA](256x1024) @ Wcat^T, grid (B/32=8, G4/128=16)
__global__ void __launch_bounds__(256)
k_lstm_step(const _Float16* __restrict__ xA,    // (B,HSZ) input half of K
            const _Float16* __restrict__ hA,    // (B,HSZ) hidden half of K
            const _Float16* __restrict__ Wcat,  // (G4,KCAT) permuted, row-major
            const float* __restrict__ biascat,  // (G4) permuted fused bias
            float* __restrict__ c,              // (B,HSZ)
            _Float16* __restrict__ houtA,       // (B,HSZ)
            _Float16* __restrict__ houtB,       // nullable
            float* __restrict__ hout32) {       // nullable
    __shared__ __align__(32) unsigned char smem[SMEM_BYTES];
    _Float16* Ab = (_Float16*)smem;             // [2][32][LDT]
    _Float16* Bb = Ab + 2 * A_EL;               // [2][128][LDT]

    const int tid  = threadIdx.x;
    const int wave = tid >> 5;
    const int lane = tid & 31;
    const int wr = wave & 1, wc = wave >> 1;
    const int r  = lane & 15, kh = lane >> 4;
    const int m0blk = blockIdx.x * 32;
    const int n0blk = blockIdx.y * 128;

    // staging assignments: A tile 32x64 halves -> 1 x 16B chunk / thread,
    //                      B tile 128x64 halves -> 4 x 16B chunks / thread
    const int arow = tid >> 3, acol = (tid & 7) * 8;
    int brow[4], bcol[4];
#pragma unroll
    for (int j = 0; j < 4; ++j) {
        int cch = tid + j * 256;
        brow[j] = cch >> 3;
        bcol[j] = (cch & 7) * 8;
    }

    v8h regA;
    v8h regB[4];
    auto load_regs = [&](int kt) {
        int k0 = kt * KT;
        const _Float16* srcA = (k0 < HSZ) ? (xA + k0) : (hA + (k0 - HSZ));
        regA = *(const v8h*)(srcA + (size_t)(m0blk + arow) * HSZ + acol);
        const _Float16* srcB = Wcat + k0;
#pragma unroll
        for (int j = 0; j < 4; ++j)
            regB[j] = *(const v8h*)(srcB + (size_t)(n0blk + brow[j]) * KCAT + bcol[j]);
    };
    auto store_regs = [&](int p) {
        *(v8h*)(Ab + p * A_EL + arow * LDT + acol) = regA;
#pragma unroll
        for (int j = 0; j < 4; ++j)
            *(v8h*)(Bb + p * B_EL + brow[j] * LDT + bcol[j]) = regB[j];
    };

    v8f acc0 = {}, acc1 = {};
    auto compute = [&](int p) {
        const _Float16* aF  = Ab + p * A_EL + (wr * 16 + r) * LDT + kh * 16;
        const _Float16* b0F = Bb + p * B_EL + (wc * 32 + r) * LDT + kh * 16;
        const _Float16* b1F = b0F + 16 * LDT;
#pragma unroll
        for (int kc = 0; kc < KT; kc += 32) {
            v16h a  = *(const v16h*)(aF + kc);
            v16h b0 = *(const v16h*)(b0F + kc);
            v16h b1 = *(const v16h*)(b1F + kc);
            acc0 = __builtin_amdgcn_wmma_f32_16x16x32_f16(false, a, false, b0, (short)0, acc0, false, false);
            acc1 = __builtin_amdgcn_wmma_f32_16x16x32_f16(false, a, false, b1, (short)0, acc1, false, false);
        }
    };

    // software pipeline: prefetch to regs, compute current LDS tile, store, barrier
    load_regs(0);
    store_regs(0);
    __syncthreads();
    constexpr int NKT = KCAT / KT;   // 16
#pragma unroll 1
    for (int kt = 0; kt < NKT; ++kt) {
        int p = kt & 1;
        if (kt + 1 < NKT) load_regs(kt + 1);
        compute(p);
        if (kt + 1 < NKT) {
            store_regs(p ^ 1);
            __syncthreads();
        }
    }
    __syncthreads();   // done with staging LDS; gate tile aliases it below

    // gates tile 32 x 128 f32 (padded stride 132), aliased over smem
    float* gt = (float*)smem;
#pragma unroll
    for (int v = 0; v < 8; ++v) {
        gt[(wr * 16 + kh * 8 + v) * 132 + wc * 32 + r]      = acc0[v];
        gt[(wr * 16 + kh * 8 + v) * 132 + wc * 32 + 16 + r] = acc1[v];
    }
    __syncthreads();

    // cell update: 32 rows x 32 hidden units, gates interleaved [i|f|g|o] x 32
    const float* bia = biascat + n0blk;
    for (int idx = tid; idx < 1024; idx += 256) {
        int rr = idx >> 5, u = idx & 31;
        int h  = blockIdx.y * 32 + u;
        int b  = m0blk + rr;
        float gi = gt[rr * 132 + u]      + bia[u];
        float gf = gt[rr * 132 + 32 + u] + bia[32 + u];
        float gg = gt[rr * 132 + 64 + u] + bia[64 + u];
        float go = gt[rr * 132 + 96 + u] + bia[96 + u];
        size_t o = (size_t)b * HSZ + h;
        float cn = sigf(gf) * c[o] + sigf(gi) * tanhf(gg);
        float hn = sigf(go) * tanhf(cn);
        c[o] = cn;
        houtA[o] = (_Float16)hn;
        if (houtB)  houtB[o]  = (_Float16)hn;
        if (hout32) hout32[o] = hn;
    }
}

// ---------------------------------------------------------------------------
// conv_vecs flat [b][f*HSZ+h] = relu( sum_t relu(hs[t][b][h]) * cw[f][t] + convb[f] )
__global__ void k_conv(const float* __restrict__ hs, const float* __restrict__ convW,
                       const float* __restrict__ convb, float* __restrict__ cv) {
    int i = blockIdx.x * blockDim.x + threadIdx.x;
    if (i >= BB * NFN * HSZ) return;
    int b   = i / (NFN * HSZ);
    int rem = i % (NFN * HSZ);
    int f   = rem / HSZ;
    int h   = rem % HSZ;
    float s = convb[f];
    const float* hp = hs + (size_t)b * HSZ + h;
    for (int t = 0; t < TT - 1; ++t)
        s += fmaxf(hp[(size_t)t * BB * HSZ], 0.f) * convW[f * (TT - 1) + t];
    cv[i] = fmaxf(s, 0.f);
}

__global__ void k_httP(const float* __restrict__ htt, const float* __restrict__ W1,
                       const float* __restrict__ b1, float* __restrict__ httP) {
    int i = blockIdx.x * blockDim.x + threadIdx.x;
    if (i >= BB * NFN) return;
    int b = i / NFN, f = i % NFN;
    float s = b1[f];
    for (int h = 0; h < HSZ; ++h) s += htt[(size_t)b * HSZ + h] * W1[f * HSZ + h];
    httP[i] = s;
}

// alpha[b][h] = sigmoid( sum_f cv_flat[b][h*NFN+f] * httP[b][f] )   (raw-reshape view)
__global__ void k_alpha(const float* __restrict__ cv, const float* __restrict__ httP,
                        float* __restrict__ alpha) {
    int i = blockIdx.x * blockDim.x + threadIdx.x;
    if (i >= BB * HSZ) return;
    int b = i / HSZ, h = i % HSZ;
    float s = 0.f;
    const float* cp = cv + (size_t)b * NFN * HSZ + (size_t)h * NFN;
#pragma unroll
    for (int f = 0; f < NFN; ++f) s += cp[f] * httP[b * NFN + f];
    alpha[i] = sigf(s);
}

// v[b][f] = sum_h cv_flat[b][f*HSZ+h] * alpha[b][h]
__global__ void k_vvec(const float* __restrict__ cv, const float* __restrict__ alpha,
                       float* __restrict__ vb) {
    int i = blockIdx.x * blockDim.x + threadIdx.x;
    if (i >= BB * NFN) return;
    int b = i / NFN, f = i % NFN;
    float s = 0.f;
    const float* cp = cv + (size_t)b * NFN * HSZ + (size_t)f * HSZ;
    for (int h = 0; h < HSZ; ++h) s += cp[h] * alpha[(size_t)b * HSZ + h];
    vb[i] = s;
}

// new_ht[b][j] = [htt | v] @ W2^T + b2
__global__ void k_newht(const float* __restrict__ htt, const float* __restrict__ vb,
                        const float* __restrict__ W2, const float* __restrict__ b2,
                        float* __restrict__ nh) {
    int i = blockIdx.x * blockDim.x + threadIdx.x;
    if (i >= BB * HSZ) return;
    int b = i / HSZ, j = i % HSZ;
    float s = b2[j];
    const float* w = W2 + (size_t)j * (HSZ + NFN);
    for (int k = 0; k < HSZ; ++k) s += htt[(size_t)b * HSZ + k] * w[k];
#pragma unroll
    for (int k = 0; k < NFN; ++k) s += vb[b * NFN + k] * w[HSZ + k];
    nh[i] = s;
}

__global__ void k_out(const float* __restrict__ nh, const float* __restrict__ Wout,
                      const float* __restrict__ bout, float* __restrict__ out) {
    int i = blockIdx.x * blockDim.x + threadIdx.x;
    if (i >= BB * NOUT) return;
    int b = i / NOUT, o = i % NOUT;
    float s = bout[o];
    for (int j = 0; j < HSZ; ++j) s += nh[(size_t)b * HSZ + j] * Wout[o * HSZ + j];
    out[i] = s;
}

// ---------------------------------------------------------------------------
extern "C" void kernel_launch(void* const* d_in, const int* in_sizes, int n_in,
                              void* d_out, int out_size, void* d_ws, size_t ws_size,
                              hipStream_t stream) {
    (void)in_sizes; (void)n_in; (void)out_size; (void)ws_size;

    const float* x     = (const float*)d_in[0];
    const float* W_hid = (const float*)d_in[1];
    const float* b_hid = (const float*)d_in[2];
    const float* Wih   = (const float*)d_in[3];
    const float* Whh   = (const float*)d_in[4];
    const float* bih   = (const float*)d_in[5];
    const float* bhh   = (const float*)d_in[6];
    const float* convW = (const float*)d_in[7];
    const float* convb = (const float*)d_in[8];
    const float* W1    = (const float*)d_in[9];
    const float* b1    = (const float*)d_in[10];
    const float* W2    = (const float*)d_in[11];
    const float* b2    = (const float*)d_in[12];
    const float* Wout  = (const float*)d_in[13];
    const float* bout  = (const float*)d_in[14];
    float* out = (float*)d_out;

    // Workspace carve-out (256B aligned slots)
    char* ws = (char*)d_ws;
    size_t off = 0;
    auto alloc = [&](size_t bytes) -> void* {
        void* p = ws + off;
        off = (off + bytes + 255) & ~(size_t)255;
        return p;
    };
    _Float16* x16    = (_Float16*)alloc((size_t)BB * TT * FIN * 2);
    _Float16* wh16   = (_Float16*)alloc((size_t)HSZ * FIN * 2);
    _Float16* xc16   = (_Float16*)alloc((size_t)TT * BB * HSZ * 2);
    _Float16* wcat16 = (_Float16*)alloc((size_t)NLAY * G4 * KCAT * 2);
    float*    biasc  = (float*)   alloc((size_t)NLAY * G4 * 4);
    _Float16* h0buf  = (_Float16*)alloc((size_t)2 * BB * HSZ * 2);   // parity double buffer
    _Float16* h1buf  = (_Float16*)alloc((size_t)2 * BB * HSZ * 2);
    _Float16* h0out  = (_Float16*)alloc((size_t)BB * HSZ * 2);       // layer0 -> layer1 feed
    float*    cbuf   = (float*)   alloc((size_t)NLAY * BB * HSZ * 4);
    float*    hs     = (float*)   alloc((size_t)TT * BB * HSZ * 4);  // last-layer history
    float*    cv     = (float*)   alloc((size_t)BB * NFN * HSZ * 4);
    float*    httP   = (float*)   alloc((size_t)BB * NFN * 4);
    float*    alpha  = (float*)   alloc((size_t)BB * HSZ * 4);
    float*    vb     = (float*)   alloc((size_t)BB * NFN * 4);
    float*    nh     = (float*)   alloc((size_t)BB * HSZ * 4);

    // 1) convert / pack weights & inputs to f16 (permuted gate order)
    {
        int n = BB * TT * FIN;
        k_cvt_f16<<<(n + 255) / 256, 256, 0, stream>>>(x, x16, n);
        n = HSZ * FIN;
        k_cvt_f16<<<(n + 255) / 256, 256, 0, stream>>>(W_hid, wh16, n);
        n = NLAY * G4 * KCAT;
        k_build_wcat<<<(n + 255) / 256, 256, 0, stream>>>(Wih, Whh, wcat16, n);
        n = NLAY * G4;
        k_build_bias<<<(n + 255) / 256, 256, 0, stream>>>(bih, bhh, biasc, n);
    }

    // 2) zero initial h, c
    hipMemsetAsync(h0buf, 0, (size_t)2 * BB * HSZ * 2, stream);
    hipMemsetAsync(h1buf, 0, (size_t)2 * BB * HSZ * 2, stream);
    hipMemsetAsync(cbuf,  0, (size_t)NLAY * BB * HSZ * 4, stream);

    // 3) xconcat GEMM (WMMA): (T*B, HSZ) <- x @ W_hid^T + b_hid
    {
        dim3 grid((TT * BB) / 128, HSZ / 64);
        k_xc_gemm<<<grid, 256, 0, stream>>>(x16, wh16, b_hid, xc16);
    }

    // 4) sequential LSTM recurrence, LDS-staged WMMA GEMM + fused cell update
    {
        dim3 grid(BB / 32, G4 / 128);
        const size_t BH = (size_t)BB * HSZ;
        for (int t = 0; t < TT; ++t) {
            int p = t & 1, q = (t + 1) & 1;
            // layer 0: input = xconcat[t], hidden = h0buf[p]
            k_lstm_step<<<grid, 256, 0, stream>>>(
                xc16 + (size_t)t * BH, h0buf + (size_t)p * BH, wcat16,
                biasc, cbuf,
                h0buf + (size_t)q * BH, h0out, nullptr);
            // layer 1: input = layer0 output, hidden = h1buf[p]; record hs[t]
            k_lstm_step<<<grid, 256, 0, stream>>>(
                h0out, h1buf + (size_t)p * BH, wcat16 + (size_t)G4 * KCAT,
                biasc + G4, cbuf + BH,
                h1buf + (size_t)q * BH, nullptr, hs + (size_t)t * BH);
        }
    }

    const float* htt = hs + (size_t)(TT - 1) * BB * HSZ;

    // 5) temporal-pattern attention epilogue
    {
        int n = BB * NFN * HSZ;
        k_conv<<<(n + 255) / 256, 256, 0, stream>>>(hs, convW, convb, cv);
        n = BB * NFN;
        k_httP<<<(n + 255) / 256, 256, 0, stream>>>(htt, W1, b1, httP);
        n = BB * HSZ;
        k_alpha<<<(n + 255) / 256, 256, 0, stream>>>(cv, httP, alpha);
        n = BB * NFN;
        k_vvec<<<(n + 255) / 256, 256, 0, stream>>>(cv, alpha, vb);
        n = BB * HSZ;
        k_newht<<<(n + 255) / 256, 256, 0, stream>>>(htt, vb, W2, b2, nh);
        n = BB * NOUT;
        k_out<<<(n + 255) / 256, 256, 0, stream>>>(nh, Wout, bout, out);
    }
}
